// UnifiedExpertMoE_40209483825892
// MI455X (gfx1250) — matrix operand order repaired and checked by reading, not compiled, not run on hardware
//
#include <hip/hip_runtime.h>
#include <hip/hip_bf16.h>

typedef __attribute__((ext_vector_type(16))) _Float16 v16h;
typedef __attribute__((ext_vector_type(8)))  _Float16 v8h;
typedef __attribute__((ext_vector_type(4)))  _Float16 v4h;
typedef __attribute__((ext_vector_type(8)))  float    v8f;

#define N_EXPERTS 8
#define D_MODEL   512
#define D_INNER   2048
#define T_TOK     8192
#define N_BINS    64   // (top1, top2) expert-pair bins
#define BROW      40   // padded LDS row stride (halfs): 80B -> conflict-free

// -------------------------------------------------------------------------
// CDNA5 async global->LDS copy (16B per lane) + ASYNCcnt drain.
// VGLOBAL async encoding: vdst = LDS byte offset VGPR, vaddr = 64-bit addr.
// -------------------------------------------------------------------------
__device__ __forceinline__ void async_copy_b128(unsigned lds_byte_off,
                                                const void* gsrc)
{
  asm volatile("global_load_async_to_lds_b128 %0, %1, off"
               :: "v"(lds_byte_off), "v"(gsrc) : "memory");
}
__device__ __forceinline__ void wait_asynccnt0()
{
  asm volatile("s_wait_asynccnt 0x0" ::: "memory");
}

// -------------------------------------------------------------------------
// Zero the combo-pair histogram bins (re-done every launch: deterministic).
// -------------------------------------------------------------------------
__global__ void moe_zero_bins_kernel(int* __restrict__ cnt)
{
  if (threadIdx.x < N_BINS) cnt[threadIdx.x] = 0;
}

// -------------------------------------------------------------------------
// Gating: one wave per token. softmax over 8 logits, top-2 (first-index tie
// break like lax.top_k). Writes dense [T,8] gates pre-scaled by 1/TOP_K,
// the token's expert-pair bin id, and histograms the bins.
// -------------------------------------------------------------------------
__global__ __launch_bounds__(256)
void moe_gating_kernel(const float* __restrict__ x,
                       const float* __restrict__ gw,
                       const float* __restrict__ gb,
                       float* __restrict__ gates,
                       int* __restrict__ tops,
                       int* __restrict__ cnt)
{
  const int wave = threadIdx.x >> 5;
  const int lane = threadIdx.x & 31;
  const int t = blockIdx.x * 8 + wave;
  const float* xr = x + (size_t)t * D_MODEL;

  float acc[N_EXPERTS];
#pragma unroll
  for (int e = 0; e < N_EXPERTS; ++e) acc[e] = 0.f;

  for (int k = lane; k < D_MODEL; k += 32) {
    const float xv = xr[k];
#pragma unroll
    for (int e = 0; e < N_EXPERTS; ++e) acc[e] += xv * gw[e * D_MODEL + k];
  }
#pragma unroll
  for (int off = 16; off > 0; off >>= 1) {
#pragma unroll
    for (int e = 0; e < N_EXPERTS; ++e) acc[e] += __shfl_xor(acc[e], off, 32);
  }

  float mx = -1e30f;
#pragma unroll
  for (int e = 0; e < N_EXPERTS; ++e) { acc[e] += gb[e]; mx = fmaxf(mx, acc[e]); }
  float s = 0.f;
#pragma unroll
  for (int e = 0; e < N_EXPERTS; ++e) { acc[e] = expf(acc[e] - mx); s += acc[e]; }
  const float inv = 1.f / s;
#pragma unroll
  for (int e = 0; e < N_EXPERTS; ++e) acc[e] *= inv;

  int i1 = 0; float v1 = acc[0];
#pragma unroll
  for (int e = 1; e < N_EXPERTS; ++e) if (acc[e] > v1) { v1 = acc[e]; i1 = e; }
  int i2 = -1; float v2 = -1.f;
#pragma unroll
  for (int e = 0; e < N_EXPERTS; ++e) if (e != i1 && acc[e] > v2) { v2 = acc[e]; i2 = e; }

  if (lane == 0) {
#pragma unroll
    for (int e = 0; e < N_EXPERTS; ++e) {
      const float g = (e == i1) ? v1 : (e == i2) ? v2 : 0.f;
      gates[t * N_EXPERTS + e] = g * 0.5f;   // fold 1/TOP_K
    }
    const int bin = i1 * N_EXPERTS + i2;
    tops[t] = bin;
    atomicAdd(&cnt[bin], 1);
  }
}

// -------------------------------------------------------------------------
// Exclusive prefix over the 64 bins -> scatter cursors (tiny, single lane).
// -------------------------------------------------------------------------
__global__ void moe_prefix_kernel(const int* __restrict__ cnt,
                                  int* __restrict__ cursor)
{
  if (threadIdx.x == 0) {
    int run = 0;
    for (int i = 0; i < N_BINS; ++i) { cursor[i] = run; run += cnt[i]; }
  }
}

// -------------------------------------------------------------------------
// Scatter: counting-sort tokens by expert-pair bin -> perm[0..T).
// -------------------------------------------------------------------------
__global__ __launch_bounds__(256)
void moe_scatter_kernel(const int* __restrict__ tops,
                        int* __restrict__ cursor,
                        int* __restrict__ perm)
{
  const int t = blockIdx.x * 256 + threadIdx.x;
  const int bin = tops[t];
  const int pos = atomicAdd(&cursor[bin], 1);
  perm[pos] = t;
}

// -------------------------------------------------------------------------
// x fp32 -> f16, 4 elements per thread
// -------------------------------------------------------------------------
__global__ __launch_bounds__(256)
void moe_cvt_x_kernel(const float* __restrict__ x, _Float16* __restrict__ xh)
{
  const int i = (blockIdx.x * 256 + threadIdx.x) * 4;
  const float4 v = *(const float4*)(x + i);
  v4h h;
  h[0] = (_Float16)v.x; h[1] = (_Float16)v.y;
  h[2] = (_Float16)v.z; h[3] = (_Float16)v.w;
  *(v4h*)(xh + i) = h;
}

// -------------------------------------------------------------------------
// W[e][k][n] fp32  ->  wt[e][n][k] f16   (LDS-tiled 32x32 transpose)
// -------------------------------------------------------------------------
__global__ __launch_bounds__(256)
void moe_cvt_w_kernel(const float* __restrict__ w, _Float16* __restrict__ wtd)
{
  __shared__ _Float16 tile[32][33];
  const int e  = blockIdx.z;
  const int n0 = blockIdx.x * 32;
  const int k0 = blockIdx.y * 32;
  const int tx = threadIdx.x & 31;
  const int ty = threadIdx.x >> 5;   // 0..7
#pragma unroll
  for (int i = 0; i < 4; ++i) {
    const int k = k0 + ty + i * 8;
    tile[ty + i * 8][tx] = (_Float16)w[((size_t)e * D_MODEL + k) * D_INNER + n0 + tx];
  }
  __syncthreads();
#pragma unroll
  for (int i = 0; i < 4; ++i) {
    const int n = n0 + ty + i * 8;
    wtd[((size_t)e * D_INNER + n) * D_MODEL + k0 + tx] = tile[tx][ty + i * 8];
  }
}

// -------------------------------------------------------------------------
// Main GEMM over the expert-pair-sorted token order, with async-LDS staged B.
// Block = 128 permuted tokens x 128 d_inner, 8 waves (4 M x 2 N).
// Pipeline over (k-chunk x active-expert): stage next B tile (8KB) via
// global_load_async_to_lds_b128 while computing on the current one out of
// LDS; drain with s_wait_asynccnt + barrier. All 8 ds_load_b128 for the
// four B fragments are issued into distinct registers before the WMMA
// chain so waits retire on partial DScnt instead of stalling per pair.
// -------------------------------------------------------------------------
__global__ __launch_bounds__(256)
void moe_gemm_kernel(const _Float16* __restrict__ xh,
                     const _Float16* __restrict__ wt,
                     const float* __restrict__ gates,
                     const float* __restrict__ bias,
                     const int* __restrict__ perm,
                     float* __restrict__ out)
{
  __shared__ _Float16 sB[2][128][BROW];   // 2 x 10,240 B, 80B row stride
  __shared__ float s_g[128 * N_EXPERTS];
  __shared__ int   s_perm[128];
  __shared__ int   s_mask;

  const int tileM = blockIdx.x * 128;
  const int tileN = blockIdx.y * 128;
  const int lane  = threadIdx.x & 31;
  const int wave  = threadIdx.x >> 5;
  const int wm    = wave & 3;
  const int wn    = wave >> 2;
  const int colBase = tileN + wn * 64;

  if (threadIdx.x == 0) s_mask = 0;
  __syncthreads();
  if (threadIdx.x < 128) {
    const int tok = perm[tileM + threadIdx.x];
    s_perm[threadIdx.x] = tok;
    int m = 0;
#pragma unroll
    for (int e = 0; e < N_EXPERTS; ++e) {
      const float g = gates[tok * N_EXPERTS + e];
      s_g[threadIdx.x * N_EXPERTS + e] = g;
      if (g != 0.f) m |= (1 << e);
    }
    atomicOr(&s_mask, m);
  }
  __syncthreads();
  const int mask = s_mask;

  // uniform active-expert list (sorted tiles: usually 2 entries)
  int elist[N_EXPERTS];
  int ne = 0;
#pragma unroll
  for (int e = 0; e < N_EXPERTS; ++e)
    if ((mask >> e) & 1) elist[ne++] = e;

  // per-lane gathered rows + gate scalars (f16) for the two 16-row A blocks
  const int rl0 = wm * 32 + (lane & 15);
  const int rl1 = rl0 + 16;
  const int tok0 = s_perm[rl0];
  const int tok1 = s_perm[rl1];
  _Float16 gh0[N_EXPERTS], gh1[N_EXPERTS];
#pragma unroll
  for (int e = 0; e < N_EXPERTS; ++e) {
    gh0[e] = (_Float16)s_g[rl0 * N_EXPERTS + e];
    gh1[e] = (_Float16)s_g[rl1 * N_EXPERTS + e];
  }

  v8f c[2][4];
#pragma unroll
  for (int i = 0; i < 2; ++i)
#pragma unroll
    for (int j = 0; j < 4; ++j) { v8f z = {}; c[i][j] = z; }

  // 16-bit A layout: lanes<16 hold K 0-7 / 16-23, lanes>=16 hold K 8-15 / 24-31
  const int kaOff = (lane & 16) ? 8 : 0;
  // 16-bit B layout: lanes<16 hold K 0-7 / 16-23, lanes>=16 hold K 8-15 / 24-31
  const int kbOff = (lane & 16) ? 8 : 0;
  const _Float16* a0p = xh + (size_t)tok0 * D_MODEL + kaOff;
  const _Float16* a1p = xh + (size_t)tok1 * D_MODEL + kaOff;

  // --- B tile staging: 128 n-rows x 32 k-halfs (64B) = 512 x 16B chunks,
  //     2 chunks per thread.  LDS row padded to BROW halfs.
  const int chunk0 = threadIdx.x * 2;
  const int sn0 = chunk0 >> 2;            // n row of 1st chunk
  const int sk0 = (chunk0 & 3) * 8;       // k-half offset of 1st chunk
  const int sn1 = (chunk0 + 1) >> 2;
  const int sk1 = ((chunk0 + 1) & 3) * 8;
  const unsigned ldsOff0 = (unsigned)(uintptr_t)&sB[0][sn0][sk0];
  const unsigned ldsOff1 = (unsigned)(uintptr_t)&sB[0][sn1][sk1];
  const unsigned bufStride =
      (unsigned)((uintptr_t)&sB[1][0][0] - (uintptr_t)&sB[0][0][0]);

#define STAGE_B(buf, e, kc)                                                   \
  do {                                                                        \
    const _Float16* base_ = wt + ((size_t)(e) * D_INNER + tileN) * D_MODEL +  \
                            (kc) * 32;                                        \
    async_copy_b128(ldsOff0 + (buf) * bufStride,                              \
                    base_ + (size_t)sn0 * D_MODEL + sk0);                     \
    async_copy_b128(ldsOff1 + (buf) * bufStride,                              \
                    base_ + (size_t)sn1 * D_MODEL + sk1);                     \
  } while (0)

  // prologue: stage (kc=0, first expert) into buffer 0
  STAGE_B(0, elist[0], 0);

  const _Float16* bbase = &sB[0][wn * 64 + (lane & 15)][kbOff];

  int it = 0;
  for (int kc = 0; kc < D_MODEL / 32; ++kc) {
    // A fragments for this k-chunk (latency hidden behind wait+barrier)
    v16h a0, a1;
    {
      const v8h lo0 = *(const v8h*)(a0p + kc * 32);
      const v8h hi0 = *(const v8h*)(a0p + kc * 32 + 16);
      const v8h lo1 = *(const v8h*)(a1p + kc * 32);
      const v8h hi1 = *(const v8h*)(a1p + kc * 32 + 16);
#pragma unroll
      for (int i = 0; i < 8; ++i) {
        a0[i] = lo0[i]; a0[i + 8] = hi0[i];
        a1[i] = lo1[i]; a1[i + 8] = hi1[i];
      }
    }
    for (int ei = 0; ei < ne; ++ei) {
      const int cur = it & 1;
      wait_asynccnt0();
      __syncthreads();          // sB[cur] ready; prev reads done (WAR safe)

      // stage the next (kc, expert) tile into the other buffer
      if (ei + 1 < ne)            STAGE_B(cur ^ 1, elist[ei + 1], kc);
      else if (kc + 1 < D_MODEL / 32) STAGE_B(cur ^ 1, elist[0], kc + 1);

      // issue all 8 ds_load_b128 into distinct regs up front
      const _Float16* bb = bbase + (size_t)cur * bufStride / 2;
      const v8h b0l = *(const v8h*)(bb + (size_t)( 0) * BROW);
      const v8h b0h = *(const v8h*)(bb + (size_t)( 0) * BROW + 16);
      const v8h b1l = *(const v8h*)(bb + (size_t)(16) * BROW);
      const v8h b1h = *(const v8h*)(bb + (size_t)(16) * BROW + 16);
      const v8h b2l = *(const v8h*)(bb + (size_t)(32) * BROW);
      const v8h b2h = *(const v8h*)(bb + (size_t)(32) * BROW + 16);
      const v8h b3l = *(const v8h*)(bb + (size_t)(48) * BROW);
      const v8h b3h = *(const v8h*)(bb + (size_t)(48) * BROW + 16);
      v16h b0, b1, b2, b3;
#pragma unroll
      for (int i = 0; i < 8; ++i) {
        b0[i] = b0l[i]; b0[i + 8] = b0h[i];
        b1[i] = b1l[i]; b1[i + 8] = b1h[i];
        b2[i] = b2l[i]; b2[i + 8] = b2h[i];
        b3[i] = b3l[i]; b3[i + 8] = b3h[i];
      }

      const int e = elist[ei];
      v16h as0, as1;
#pragma unroll
      for (int i = 0; i < 16; ++i) {
        as0[i] = a0[i] * gh0[e];
        as1[i] = a1[i] * gh1[e];
      }

      c[0][0] = __builtin_amdgcn_wmma_f32_16x16x32_f16(
          false, as0, false, b0, (short)0, c[0][0], false, false);
      c[1][0] = __builtin_amdgcn_wmma_f32_16x16x32_f16(
          false, as1, false, b0, (short)0, c[1][0], false, false);
      c[0][1] = __builtin_amdgcn_wmma_f32_16x16x32_f16(
          false, as0, false, b1, (short)0, c[0][1], false, false);
      c[1][1] = __builtin_amdgcn_wmma_f32_16x16x32_f16(
          false, as1, false, b1, (short)0, c[1][1], false, false);
      c[0][2] = __builtin_amdgcn_wmma_f32_16x16x32_f16(
          false, as0, false, b2, (short)0, c[0][2], false, false);
      c[1][2] = __builtin_amdgcn_wmma_f32_16x16x32_f16(
          false, as1, false, b2, (short)0, c[1][2], false, false);
      c[0][3] = __builtin_amdgcn_wmma_f32_16x16x32_f16(
          false, as0, false, b3, (short)0, c[0][3], false, false);
      c[1][3] = __builtin_amdgcn_wmma_f32_16x16x32_f16(
          false, as1, false, b3, (short)0, c[1][3], false, false);
      ++it;
    }
  }
#undef STAGE_B

  // epilogue: out[tok] = C + sum_e g[tok,e] * bias[e][n]  (gates already /K)
#pragma unroll
  for (int nb = 0; nb < 4; ++nb) {
    const int col = colBase + nb * 16 + (lane & 15);
    float bcol[N_EXPERTS];
#pragma unroll
    for (int e = 0; e < N_EXPERTS; ++e) bcol[e] = bias[e * D_INNER + col];
#pragma unroll
    for (int mb = 0; mb < 2; ++mb) {
#pragma unroll
      for (int v = 0; v < 8; ++v) {
        const int rl = wm * 32 + mb * 16 + ((lane & 16) ? v + 8 : v);
        float bsum = 0.f;
#pragma unroll
        for (int e = 0; e < N_EXPERTS; ++e) bsum += s_g[rl * N_EXPERTS + e] * bcol[e];
        out[(size_t)s_perm[rl] * D_INNER + col] = c[mb][nb][v] + bsum;
      }
    }
  }
}

// -------------------------------------------------------------------------
extern "C" void kernel_launch(void* const* d_in, const int* in_sizes, int n_in,
                              void* d_out, int out_size, void* d_ws, size_t ws_size,
                              hipStream_t stream)
{
  const float* x  = (const float*)d_in[0];   // sequences      [4,2048,512]
  const float* w  = (const float*)d_in[1];   // expert_weights [8,512,2048]
  const float* b  = (const float*)d_in[2];   // expert_biases  [8,2048]
  const float* gw = (const float*)d_in[3];   // gating_w       [8,512]
  const float* gb = (const float*)d_in[4];   // gating_b       [8]
  float* out = (float*)d_out;                // [8192,2048] f32

  char* ws = (char*)d_ws;
  size_t off = 0;
  float* gates = (float*)(ws + off); off += (size_t)T_TOK * N_EXPERTS * 4; // 262144
  int*   tops  = (int*)(ws + off);   off += (size_t)T_TOK * 4;             //  32768
  int*   perm  = (int*)(ws + off);   off += (size_t)T_TOK * 4;             //  32768
  int*   cnt   = (int*)(ws + off);   off += N_BINS * 4;                    //    256
  int*   cursor= (int*)(ws + off);   off += N_BINS * 4;                    //    256
  off = (off + 255) & ~(size_t)255;
  _Float16* xh  = (_Float16*)(ws + off); off += (size_t)T_TOK * D_MODEL * 2;
  _Float16* wtd = (_Float16*)(ws + off); // 8*2048*512*2 = 16,777,216

  moe_zero_bins_kernel<<<1, 64, 0, stream>>>(cnt);
  moe_gating_kernel<<<T_TOK / 8, 256, 0, stream>>>(x, gw, gb, gates, tops, cnt);
  moe_prefix_kernel<<<1, 32, 0, stream>>>(cnt, cursor);
  moe_scatter_kernel<<<T_TOK / 256, 256, 0, stream>>>(tops, cursor, perm);
  moe_cvt_x_kernel<<<(T_TOK * D_MODEL) / (256 * 4), 256, 0, stream>>>(x, xh);
  moe_cvt_w_kernel<<<dim3(D_INNER / 32, D_MODEL / 32, N_EXPERTS), 256, 0, stream>>>(w, wtd);
  moe_gemm_kernel<<<dim3(T_TOK / 128, D_INNER / 128), 256, 0, stream>>>(
      xh, wtd, gates, b, perm, out);
}
